// GlobalAttn_14826227106230
// MI455X (gfx1250) — matrix operand order repaired and checked by reading, not compiled, not run on hardware
//
#include <hip/hip_runtime.h>
#include <hip/hip_bf16.h>

typedef __bf16 bf16;
typedef __attribute__((ext_vector_type(16))) __bf16 v16bf;
typedef __attribute__((ext_vector_type(8)))  __bf16 v8bf;
typedef __attribute__((ext_vector_type(8)))  float  v8f;

static constexpr int BB  = 2;
static constexpr int SEQ = 2048;
static constexpr int EMB = 1024;
static constexpr int NH  = 16;
static constexpr int DH  = 64;

union V16 { v16bf v; v8bf h[2]; };

__device__ inline v8f wmma_bf16(v16bf a, v16bf b, v8f c) {
  return __builtin_amdgcn_wmma_f32_16x16x32_bf16(
      /*neg_a=*/false, a, /*neg_b=*/false, b,
      /*c_mod=*/(short)0, c, /*reuse_a=*/false, /*reuse_b=*/false);
}

// A-matrix fragment (16x32 bf16, row-major source, leading dim ld):
// lanes 0-15 (row m) hold K = {k0..k0+7, k0+16..k0+23}; lanes 16-31 the other halves.
__device__ inline v16bf load_a(const bf16* base, int ld, int m0, int k0, int lane) {
  int m = lane & 15, hi = lane >> 4;
  const bf16* p = base + (size_t)(m0 + m) * ld + k0 + hi * 8;
  V16 u;
  u.h[0] = *(const v8bf*)(p);
  u.h[1] = *(const v8bf*)(p + 16);
  return u.v;
}

// B-matrix fragment (32x16 bf16). Source is B^T row-major (row n = column n of B):
// lanes 0-15 hold K = {k0..k0+7, k0+16..k0+23} of column n; lanes 16-31 the other halves.
__device__ inline v16bf load_b(const bf16* base, int ld, int n0, int k0, int lane) {
  int n = lane & 15, hi = lane >> 4;
  const bf16* p = base + (size_t)(n0 + n) * ld + k0 + hi * 8;
  V16 u;
  u.h[0] = *(const v8bf*)(p);
  u.h[1] = *(const v8bf*)(p + 16);
  return u.v;
}

// ---------------- f32 -> bf16 conversion ----------------
__global__ void k_cvt(const float* __restrict__ in, bf16* __restrict__ out, int n) {
  int i = blockIdx.x * blockDim.x + threadIdx.x;
  if (i < n) out[i] = (bf16)in[i];
}

// ---------------- QKV projection: y[b,s,f] = sum_e x[b,s,e] * W[f,e] ----------------
// Grid: (B*S/128, E/64, 3); block 256 (8 waves). Wave w: 16 rows x 64 cols.
// z=0 -> q[b,h,s,d], z=1 -> k[b,h,s,d], z=2 -> vT[b,h,d,s] (all bf16).
__global__ __launch_bounds__(256) void k_proj(
    const bf16* __restrict__ xh, const bf16* __restrict__ Wq,
    const bf16* __restrict__ Wk, const bf16* __restrict__ Wv,
    bf16* __restrict__ qh, bf16* __restrict__ kh, bf16* __restrict__ vT) {
  int lane = threadIdx.x & 31, w = threadIdx.x >> 5;
  int row0 = blockIdx.x * 128 + w * 16;   // global row over B*S
  int f0   = blockIdx.y * 64;             // exactly one head's d-range
  int z    = blockIdx.z;
  const bf16* W = (z == 0) ? Wq : ((z == 1) ? Wk : Wv);

  v8f acc[4] = {};
  for (int e0 = 0; e0 < EMB; e0 += 32) {
    v16bf a = load_a(xh, EMB, row0, e0, lane);
#pragma unroll
    for (int t = 0; t < 4; ++t) {
      v16bf b = load_b(W, EMB, f0 + 16 * t, e0, lane);   // rows of W = cols of W^T
      acc[t] = wmma_bf16(a, b, acc[t]);
    }
  }

  int bcol = lane & 15, hi = lane >> 4;
  int rbase = row0 + 8 * hi;
  int bidx  = rbase >> 11;          // batch (SEQ = 2048)
  int sbase = rbase & (SEQ - 1);
#pragma unroll
  for (int t = 0; t < 4; ++t) {
    int f  = f0 + 16 * t + bcol;
    int hh = f >> 6, dd = f & 63;
    if (z < 2) {
      bf16* dst = (z == 0) ? qh : kh;
#pragma unroll
      for (int i = 0; i < 8; ++i)
        dst[((size_t)(bidx * NH + hh) * SEQ + (sbase + i)) * DH + dd] = (bf16)acc[t][i];
    } else {
      v8bf pv;
#pragma unroll
      for (int i = 0; i < 8; ++i) pv[i] = (bf16)acc[t][i];
      *(v8bf*)(vT + ((size_t)(bidx * NH + hh) * DH + dd) * SEQ + sbase) = pv;
    }
  }
}

// ---------------- Flash attention with relative-position skew ----------------
// Grid: (S/64, B*H); block 128 (4 waves). Wave w: 16 query rows.
// K/V/Er tiles are staged once per block into LDS (shared by all 4 waves).
__global__ __launch_bounds__(128) void k_attn(
    const bf16* __restrict__ qh, const bf16* __restrict__ kh,
    const bf16* __restrict__ vT, const bf16* __restrict__ Erh,
    float* __restrict__ out) {
  // ld = 80 halves (160B pitch): keeps ds_read_b128 16B-aligned, spreads banks.
  __shared__ __align__(16) bf16 Kt[64 * 80];    // K tile  [key][d]
  __shared__ __align__(16) bf16 Vt[64 * 80];    // V tile  [d][key]
  __shared__ __align__(16) bf16 Et[128 * 80];   // Er rows [r-R0][d]
  __shared__ __align__(16) bf16 Pl[4][16 * 80]; // per-wave P tiles

  int tid  = threadIdx.x;
  int lane = tid & 31, w = tid >> 5;
  int bh = blockIdx.y;
  int b = bh >> 4, h = bh & 15;
  int q0b = blockIdx.x * 64;
  int q0  = q0b + w * 16;
  const bf16* qp = qh + (size_t)bh * SEQ * DH;
  const bf16* kp = kh + (size_t)bh * SEQ * DH;
  const bf16* vp = vT + (size_t)bh * DH * SEQ;
  int bcol = lane & 15, hi = lane >> 4;

  v16bf aq0 = load_a(qp, DH, q0, 0, lane);
  v16bf aq1 = load_a(qp, DH, q0, 32, lane);

  v8f o[4] = {};
  float mrow[8], lrow[8];
#pragma unroll
  for (int i = 0; i < 8; ++i) { mrow[i] = -1e30f; lrow[i] = 0.f; }

  bf16* Pw = &Pl[w][0];
  int srow = tid >> 1;          // staging row 0..63
  int scol = (tid & 1) * 32;    // staging col 0 or 32

  for (int kb = 0; kb <= (int)blockIdx.x; ++kb) {
    int k0g = kb * 64;

    // ---- cooperative staging: K (64x64), V^T (64x64), Er window (128x64) ----
    {
      const bf16* gK = kp + (size_t)(k0g + srow) * DH + scol;
      const bf16* gV = vp + (size_t)srow * SEQ + k0g + scol;
#pragma unroll
      for (int j = 0; j < 4; ++j) {
        *(v8bf*)&Kt[srow * 80 + scol + 8 * j] = *(const v8bf*)(gK + 8 * j);
        *(v8bf*)&Vt[srow * 80 + scol + 8 * j] = *(const v8bf*)(gV + 8 * j);
      }
      int R0 = k0g + (SEQ - 16) - (q0b + 48);   // min rel row over all waves
      int r = R0 + tid;
      r = r < 0 ? 0 : (r > SEQ - 1 ? SEQ - 1 : r);
      const bf16* gE = Erh + (size_t)r * DH;
#pragma unroll
      for (int j = 0; j < 8; ++j)
        *(v8bf*)&Et[tid * 80 + 8 * j] = *(const v8bf*)(gE + 8 * j);
    }
    // prefetch next key block's K/V tiles into cache while we compute
    if (kb < (int)blockIdx.x) {
      int nk = k0g + 64;
      __builtin_prefetch(kp + (size_t)(nk + srow) * DH + scol, 0, 0);
      __builtin_prefetch(vp + (size_t)srow * SEQ + nk + scol, 0, 0);
    }
    __syncthreads();

    // ---- S = q . k^T (16x64) from LDS ----
    v8f s[4] = {};
#pragma unroll
    for (int t = 0; t < 4; ++t) {
      s[t] = wmma_bf16(aq0, load_b(Kt, 80, 16 * t, 0, lane), s[t]);
      s[t] = wmma_bf16(aq1, load_b(Kt, 80, 16 * t, 32, lane), s[t]);
    }

    // ---- Srel: this wave's rel rows live at local offset (48-16w)+j ----
    int eoff = 48 - 16 * w;
    v8f e[5] = {};
#pragma unroll
    for (int t = 0; t < 5; ++t) {
      e[t] = wmma_bf16(aq0, load_b(Et, 80, eoff + 16 * t, 0, lane), e[t]);
      e[t] = wmma_bf16(aq1, load_b(Et, 80, eoff + 16 * t, 32, lane), e[t]);
    }

    // Combine, scale, causal mask. Element (a,b) of subtile t needs E[a, 16t+15-a+b]:
    // a row-preserving cross-lane rotation inside each 16-lane half.
#pragma unroll
    for (int t = 0; t < 4; ++t) {
#pragma unroll
      for (int i = 0; i < 8; ++i) {
        int a = i + 8 * hi;
        int j = 16 * t + 15 - a + bcol;            // in [16t, 16t+30]
        int src = (j & 15) | (hi << 4);
        float vlo = __shfl(e[t][i], src, 32);
        float vhi = __shfl(e[t + 1][i], src, 32);
        float rel = (j < 16 * (t + 1)) ? vlo : vhi;
        float sv = (s[t][i] + rel) * 0.125f;       // 1/sqrt(64)
        int ig = q0 + a, kg = k0g + 16 * t + bcol;
        s[t][i] = (kg <= ig) ? sv : -1e30f;
      }
    }

    // ---- online softmax (row = (i,hi); columns across the 16-lane half) ----
#pragma unroll
    for (int i = 0; i < 8; ++i) {
      float mx = fmaxf(fmaxf(s[0][i], s[1][i]), fmaxf(s[2][i], s[3][i]));
#pragma unroll
      for (int off = 1; off < 16; off <<= 1) mx = fmaxf(mx, __shfl_xor(mx, off, 32));
      float mnew  = fmaxf(mrow[i], mx);
      float alpha = __expf(mrow[i] - mnew);
      float rs = 0.f;
#pragma unroll
      for (int t = 0; t < 4; ++t) { float p = __expf(s[t][i] - mnew); s[t][i] = p; rs += p; }
#pragma unroll
      for (int off = 1; off < 16; off <<= 1) rs += __shfl_xor(rs, off, 32);
      lrow[i] = lrow[i] * alpha + rs;
      mrow[i] = mnew;
#pragma unroll
      for (int t = 0; t < 4; ++t) o[t][i] *= alpha;
    }

    // ---- P tile (C-layout) -> LDS bf16; same-wave DS ops are in-order ----
#pragma unroll
    for (int t = 0; t < 4; ++t)
#pragma unroll
      for (int i = 0; i < 8; ++i)
        Pw[(i + 8 * hi) * 80 + 16 * t + bcol] = (bf16)s[t][i];

    // ---- O += P . V from LDS ----
#pragma unroll
    for (int kk = 0; kk < 64; kk += 32) {
      v16bf ap = load_a(Pw, 80, 0, kk, lane);
#pragma unroll
      for (int t = 0; t < 4; ++t) {
        v16bf bv = load_b(Vt, 80, 16 * t, kk, lane);  // V^T rows = V columns
        o[t] = wmma_bf16(ap, bv, o[t]);
      }
    }
    __syncthreads();   // protect staged tiles before next iteration (uniform trips)
  }

  // ---- normalize and emit out[b, s, h*64 + d] (f32) ----
#pragma unroll
  for (int t = 0; t < 4; ++t) {
#pragma unroll
    for (int i = 0; i < 8; ++i) {
      int sg = q0 + i + 8 * hi;
      out[((size_t)b * SEQ + sg) * EMB + h * DH + 16 * t + bcol] = o[t][i] / lrow[i];
    }
  }
}

extern "C" void kernel_launch(void* const* d_in, const int* in_sizes, int n_in,
                              void* d_out, int out_size, void* d_ws, size_t ws_size,
                              hipStream_t stream) {
  const float* x  = (const float*)d_in[0];
  const float* Wq = (const float*)d_in[1];
  const float* Wk = (const float*)d_in[2];
  const float* Wv = (const float*)d_in[3];
  const float* Er = (const float*)d_in[4];   // S == MAX_LEN -> Er_s == Er

  char* ws = (char*)d_ws;
  size_t off = 0;
  bf16* xh  = (bf16*)(ws + off); off += (size_t)BB * SEQ * EMB * 2;   // 8 MB
  bf16* Wqh = (bf16*)(ws + off); off += (size_t)EMB * EMB * 2;        // 2 MB
  bf16* Wkh = (bf16*)(ws + off); off += (size_t)EMB * EMB * 2;
  bf16* Wvh = (bf16*)(ws + off); off += (size_t)EMB * EMB * 2;
  bf16* Erh = (bf16*)(ws + off); off += (size_t)SEQ * DH * 2;         // 256 KB
  bf16* qh  = (bf16*)(ws + off); off += (size_t)BB * NH * SEQ * DH * 2;
  bf16* kh  = (bf16*)(ws + off); off += (size_t)BB * NH * SEQ * DH * 2;
  bf16* vTh = (bf16*)(ws + off); off += (size_t)BB * NH * DH * SEQ * 2;

  auto cvt = [&](const float* src, bf16* dst, int n) {
    k_cvt<<<(n + 255) / 256, 256, 0, stream>>>(src, dst, n);
  };
  cvt(x,  xh,  BB * SEQ * EMB);
  cvt(Wq, Wqh, EMB * EMB);
  cvt(Wk, Wkh, EMB * EMB);
  cvt(Wv, Wvh, EMB * EMB);
  cvt(Er, Erh, SEQ * DH);

  dim3 gp((BB * SEQ) / 128, EMB / 64, 3);
  k_proj<<<gp, 256, 0, stream>>>(xh, Wqh, Wkh, Wvh, qh, kh, vTh);

  dim3 ga(SEQ / 64, BB * NH);
  k_attn<<<ga, 128, 0, stream>>>(qh, kh, vTh, Erh, (float*)d_out);
}
